// LlamaLayer_7550552506705
// MI455X (gfx1250) — matrix operand order repaired and checked by reading, not compiled, not run on hardware
//
#include <hip/hip_runtime.h>
#include <hip/hip_bf16.h>

// ---------------------------------------------------------------------------
// Llama layer on MI455X (gfx1250, wave32, WMMA, async LDS DMA).
//
// Roofline: ~1 TFLOP of GEMM, compute bound -> all matmuls via
// v_wmma_f32_16x16x32_f16 (f32 accum). Weights are converted f32->f16 once
// into a reused staging buffer (halves B-side HBM traffic; B tiles are
// re-read M/128 = 16x). All GEMM operands f16 -> global->LDS staging uses
// gfx1250 GLOBAL_LOAD_ASYNC_TO_LDS_B128 (ASYNCcnt) with double-buffered LDS
// so the DMA overlaps WMMA.
//
// Workspace requirement: ~568 MB (carved below).
// ---------------------------------------------------------------------------

typedef _Float16 h4   __attribute__((ext_vector_type(4)));
typedef _Float16 h8   __attribute__((ext_vector_type(8)));
typedef _Float16 v16h __attribute__((ext_vector_type(16)));
typedef float    v8f  __attribute__((ext_vector_type(8)));
typedef float    f4   __attribute__((ext_vector_type(4)));

union AFrag { v16h v; h8 h[2]; };

enum { EPI_F32 = 0, EPI_F32_RES = 1, EPI_F16 = 2 };

// Low 32 bits of a generic pointer to LDS = byte offset from LDS base
// (generic LDS addresses are {aperture, offset[31:0]} on gfx1250).
__device__ __forceinline__ unsigned lds_off32(const void* p) {
    return (unsigned)(unsigned long long)p;
}

// Per-lane async DMA: LDS[lds] <- 16 bytes at g. Tracked by ASYNCcnt.
__device__ __forceinline__ void async_b128(unsigned lds, const _Float16* g) {
    asm volatile("global_load_async_to_lds_b128 %0, %1, off"
                 :: "v"(lds), "v"(g) : "memory");
}

__device__ __forceinline__ void wait_async0() {
    asm volatile("s_wait_asynccnt 0x0" ::: "memory");
}

// ---------------------------------------------------------------------------
// WMMA GEMM:  C[M,N] = A[M,K](f16) * B[N,K](f16)^T  (+ residual)
// Block tile 128x128, K-step 32, double-buffered LDS filled by async DMA.
// 256 threads = 8 waves (2x4); each wave owns 64x32 = 4x2 16x16 fragments.
// Grid must cover M,N exactly (all problem dims are multiples of 128/32).
// ---------------------------------------------------------------------------
template <int EPI>
__global__ __launch_bounds__(256) void wmma_gemm(
    const _Float16* __restrict__ A, const _Float16* __restrict__ B,
    const float* __restrict__ Res, void* __restrict__ Cout,
    int K, int lda, int ldb, int ldc)
{
    __shared__ __align__(16) _Float16 As[2][128 * 40];   // 40-half row stride
    __shared__ __align__(16) _Float16 Bs[2][128 * 40];

    const int tid  = threadIdx.x;
    const int lane = tid & 31;
    const int wid  = tid >> 5;
    const int wm   = wid >> 2;         // 0..1
    const int wn   = wid & 3;          // 0..3
    const long bm  = (long)blockIdx.y * 128;
    const long bn  = (long)blockIdx.x * 128;
    const int lm   = lane & 15;
    const int lh   = lane >> 4;

    v8f acc[4][2];
    {
        v8f z = {};
        #pragma unroll
        for (int i = 0; i < 4; ++i)
            #pragma unroll
            for (int j = 0; j < 2; ++j) acc[i][j] = z;
    }

    // Issue one 128x32 f16 tile of A and B into LDS buffer `buf` (async).
    auto issue_tile = [&](int buf, int k0) {
        #pragma unroll
        for (int i = 0; i < 2; ++i) {
            int v   = tid + i * 256;           // 0..511
            int row = v >> 2;
            int ch  = (v & 3) * 8;
            async_b128(lds_off32(&As[buf][row * 40 + ch]),
                       A + (bm + row) * lda + k0 + ch);
        }
        #pragma unroll
        for (int i = 0; i < 2; ++i) {
            int v   = tid + i * 256;
            int row = v >> 2;
            int ch  = (v & 3) * 8;
            async_b128(lds_off32(&Bs[buf][row * 40 + ch]),
                       B + (bn + row) * ldb + k0 + ch);
        }
    };

    issue_tile(0, 0);

    for (int k0 = 0, it = 0; k0 < K; k0 += 32, ++it) {
        const int cur = it & 1;
        wait_async0();          // our async loads for buffer `cur` done
        __syncthreads();        // everyone's loads done; prev compute done
        if (k0 + 32 < K) issue_tile(cur ^ 1, k0 + 32);   // overlap with WMMA

        // ---- fragments from LDS
        AFrag a[4], b[2];
        #pragma unroll
        for (int fm = 0; fm < 4; ++fm) {
            int r = wm * 64 + fm * 16 + lm;    // M row in tile
            // A 16x32 f16: lanes 0-15: K 0-7 / 16-23; lanes 16-31: K 8-15 / 24-31
            a[fm].h[0] = *(const h8*)(&As[cur][r * 40 + lh * 8]);
            a[fm].h[1] = *(const h8*)(&As[cur][r * 40 + 16 + lh * 8]);
        }
        #pragma unroll
        for (int fn = 0; fn < 2; ++fn) {
            int r = wn * 32 + fn * 16 + lm;    // N col in tile
            // B 32x16 f16: lanes 0-15: K 0-7 / 16-23; lanes 16-31: K 8-15 / 24-31
            b[fn].h[0] = *(const h8*)(&Bs[cur][r * 40 + lh * 8]);
            b[fn].h[1] = *(const h8*)(&Bs[cur][r * 40 + 16 + lh * 8]);
        }

        #pragma unroll
        for (int fm = 0; fm < 4; ++fm)
            #pragma unroll
            for (int fn = 0; fn < 2; ++fn)
                acc[fm][fn] = __builtin_amdgcn_wmma_f32_16x16x32_f16(
                    false, a[fm].v, false, b[fn].v,
                    (short)0, acc[fm][fn], false, false);
    }

    // ---- epilogue. C 16x16 f32 layout: VGPR r -> M = r + 8*(lane/16), N = lane%16
    #pragma unroll
    for (int fm = 0; fm < 4; ++fm) {
        #pragma unroll
        for (int fn = 0; fn < 2; ++fn) {
            long col = bn + wn * 32 + fn * 16 + lm;
            #pragma unroll
            for (int r = 0; r < 8; ++r) {
                long row = bm + wm * 64 + fm * 16 + lh * 8 + r;
                float vv = acc[fm][fn][r];
                if constexpr (EPI == EPI_F32) {
                    ((float*)Cout)[row * ldc + col] = vv;
                } else if constexpr (EPI == EPI_F32_RES) {
                    ((float*)Cout)[row * ldc + col] = vv + Res[row * ldc + col];
                } else {
                    ((_Float16*)Cout)[row * ldc + col] = (_Float16)vv;
                }
            }
        }
    }
}

// ---------------------------------------------------------------------------
// f32 -> f16 bulk convert (vector x4). total must be a multiple of 1024.
// ---------------------------------------------------------------------------
__global__ __launch_bounds__(256) void f32_to_f16_v4(
    const float* __restrict__ X, _Float16* __restrict__ O)
{
    long i = ((long)blockIdx.x * 256 + threadIdx.x) * 4;
    f4 d = *(const f4*)(X + i);
    *(h4*)(O + i) = __builtin_convertvector(d, h4);
}

// ---------------------------------------------------------------------------
// RMSNorm: one block per row, f32 in -> f16 out (WMMA A operand)
// ---------------------------------------------------------------------------
__global__ __launch_bounds__(256) void rmsnorm_f16(
    const float* __restrict__ X, const float* __restrict__ W,
    _Float16* __restrict__ O, int d)
{
    __shared__ float red[256];
    const int r = blockIdx.x;
    const float* row = X + (size_t)r * d;
    float s = 0.f;
    for (int c = threadIdx.x; c < d; c += 256) { float v = row[c]; s += v * v; }
    red[threadIdx.x] = s; __syncthreads();
    for (int t = 128; t > 0; t >>= 1) {
        if (threadIdx.x < t) red[threadIdx.x] += red[threadIdx.x + t];
        __syncthreads();
    }
    const float rinv = rsqrtf(red[0] / (float)d + 1e-5f);
    _Float16* orow = O + (size_t)r * d;
    for (int c = threadIdx.x; c < d; c += 256)
        orow[c] = (_Float16)(row[c] * rinv * W[c]);
}

// ---------------------------------------------------------------------------
// RoPE: out = x*cos + swap(x)*sin ; f32 in -> f16 out. cols = heads*128.
// ---------------------------------------------------------------------------
__global__ __launch_bounds__(256) void rope_f16(
    const float* __restrict__ X, const float* __restrict__ C,
    const float* __restrict__ S, _Float16* __restrict__ O, int ncols)
{
    long i = (long)blockIdx.x * 256 + threadIdx.x;   // grid covers n*ncols exactly
    int n = (int)(i / ncols);
    int c = (int)(i % ncols);
    int d = c & 127;
    float x = X[i];
    float partner = (d & 1) ? X[i - 1] : -X[i + 1];
    O[i] = (_Float16)(x * C[n * 128 + d] + partner * S[n * 128 + d]);
}

// ---------------------------------------------------------------------------
// V transpose: V[2048, 8*128] f32 -> VT[8][128][2048] f16 (B operand of AV)
// ---------------------------------------------------------------------------
__global__ __launch_bounds__(256) void transpose_v_f16(
    const float* __restrict__ V, _Float16* __restrict__ VT)
{
    long i = (long)blockIdx.x * 256 + threadIdx.x;   // 2048*1024 exactly
    int n   = (int)(i >> 10);
    int c   = (int)(i & 1023);
    int kvh = c >> 7;
    int d   = c & 127;
    VT[((size_t)kvh * 128 + d) * 2048 + n] = (_Float16)V[i];
}

// ---------------------------------------------------------------------------
// Causal softmax over one head's score matrix: block per row. f32 -> f16 P.
// ---------------------------------------------------------------------------
__global__ __launch_bounds__(256) void softmax_causal(
    const float* __restrict__ S, _Float16* __restrict__ P, int n)
{
    __shared__ float red[256];
    const int r = blockIdx.x;
    const float scale = 0.08838834764831845f;   // 1/sqrt(128)
    const float* row = S + (size_t)r * n;
    const int valid = r + 1;

    float m = -3.4e38f;
    for (int c = threadIdx.x; c < valid; c += 256) m = fmaxf(m, row[c] * scale);
    red[threadIdx.x] = m; __syncthreads();
    for (int t = 128; t > 0; t >>= 1) {
        if (threadIdx.x < t) red[threadIdx.x] = fmaxf(red[threadIdx.x], red[threadIdx.x + t]);
        __syncthreads();
    }
    m = red[0]; __syncthreads();

    float s = 0.f;
    for (int c = threadIdx.x; c < valid; c += 256) s += expf(row[c] * scale - m);
    red[threadIdx.x] = s; __syncthreads();
    for (int t = 128; t > 0; t >>= 1) {
        if (threadIdx.x < t) red[threadIdx.x] += red[threadIdx.x + t];
        __syncthreads();
    }
    const float inv = 1.0f / red[0];

    _Float16* prow = P + (size_t)r * n;
    for (int c = threadIdx.x; c < n; c += 256)
        prow[c] = (_Float16)((c < valid) ? expf(row[c] * scale - m) * inv : 0.0f);
}

// ---------------------------------------------------------------------------
// SwiGLU: ff = silu(g) * u, f32 in -> f16 out
// ---------------------------------------------------------------------------
__global__ __launch_bounds__(256) void swiglu_f16(
    const float* __restrict__ G, const float* __restrict__ U,
    _Float16* __restrict__ O)
{
    long i = (long)blockIdx.x * 256 + threadIdx.x;   // grid covers total exactly
    float g = G[i];
    O[i] = (_Float16)((g / (1.0f + expf(-g))) * U[i]);
}

// ---------------------------------------------------------------------------
// Host driver
// ---------------------------------------------------------------------------
extern "C" void kernel_launch(void* const* d_in, const int* in_sizes, int n_in,
                              void* d_out, int out_size, void* d_ws, size_t ws_size,
                              hipStream_t stream)
{
    (void)in_sizes; (void)n_in; (void)out_size; (void)ws_size;

    constexpr int Nt   = 2048;     // tokens
    constexpr int DM   = 4096;     // d_model
    constexpr int DKV  = 1024;     // n_kv_heads * d_head
    constexpr int DF   = 14336;    // d_ffn

    const float* x     = (const float*)d_in[0];
    const float* r_cos = (const float*)d_in[1];
    const float* r_sin = (const float*)d_in[2];
    const float* anw   = (const float*)d_in[3];
    const float* wq    = (const float*)d_in[4];
    const float* wk    = (const float*)d_in[5];
    const float* wv    = (const float*)d_in[6];
    const float* wo    = (const float*)d_in[7];
    const float* fnw   = (const float*)d_in[8];
    const float* wg    = (const float*)d_in[9];
    const float* wh    = (const float*)d_in[10];
    const float* wf    = (const float*)d_in[11];

    char*  ws  = (char*)d_ws;
    size_t off = 0;
    auto carve = [&](size_t bytes) -> char* {
        char* p = ws + off;
        off += (bytes + 255) & ~(size_t)255;
        return p;
    };

    _Float16* h16    = (_Float16*)carve((size_t)Nt * DM  * 2);  //  16 MB
    float*    qf     = (float*)   carve((size_t)Nt * DM  * 4);  //  32 MB
    float*    kf     = (float*)   carve((size_t)Nt * DKV * 4);  //   8 MB
    float*    vf     = (float*)   carve((size_t)Nt * DKV * 4);  //   8 MB
    _Float16* q16    = (_Float16*)carve((size_t)Nt * DM  * 2);  //  16 MB
    _Float16* k16    = (_Float16*)carve((size_t)Nt * DKV * 2);  //   4 MB
    _Float16* vt16   = (_Float16*)carve((size_t)Nt * DKV * 2);  //   4 MB
    float*    scores = (float*)   carve((size_t)Nt * Nt  * 4);  //  16 MB (per head)
    _Float16* p16    = (_Float16*)carve((size_t)Nt * Nt  * 2);  //   8 MB (per head)
    _Float16* attn16 = (_Float16*)carve((size_t)Nt * DM  * 2);  //  16 MB
    float*    x2     = (float*)   carve((size_t)Nt * DM  * 4);  //  32 MB
    _Float16* h2_16  = (_Float16*)carve((size_t)Nt * DM  * 2);  //  16 MB
    float*    gf     = (float*)   carve((size_t)Nt * DF  * 4);  // 112 MB
    float*    uf     = (float*)   carve((size_t)Nt * DF  * 4);  // 112 MB
    _Float16* ff16   = (_Float16*)carve((size_t)Nt * DF  * 2);  //  56 MB
    _Float16* wbuf   = (_Float16*)carve((size_t)DF * DM  * 2);  // 112 MB (weight staging)

    auto cvt = [&](const float* src, long total) {   // f32 weights -> f16 staging
        f32_to_f16_v4<<<total / 1024, 256, 0, stream>>>(src, wbuf);
    };

    // ---- attention RMSNorm -> h16
    rmsnorm_f16<<<Nt, 256, 0, stream>>>(x, anw, h16, DM);

    // ---- QKV projections
    cvt(wq, (long)DM * DM);
    wmma_gemm<EPI_F32><<<dim3(DM / 128,  Nt / 128), 256, 0, stream>>>(
        h16, wbuf, nullptr, qf, DM, DM, DM, DM);
    cvt(wk, (long)DKV * DM);
    wmma_gemm<EPI_F32><<<dim3(DKV / 128, Nt / 128), 256, 0, stream>>>(
        h16, wbuf, nullptr, kf, DM, DM, DM, DKV);
    cvt(wv, (long)DKV * DM);
    wmma_gemm<EPI_F32><<<dim3(DKV / 128, Nt / 128), 256, 0, stream>>>(
        h16, wbuf, nullptr, vf, DM, DM, DM, DKV);

    // ---- RoPE (f32 -> f16) and V transpose
    rope_f16<<<(Nt * DM)  / 256, 256, 0, stream>>>(qf, r_cos, r_sin, q16, DM);
    rope_f16<<<(Nt * DKV) / 256, 256, 0, stream>>>(kf, r_cos, r_sin, k16, DKV);
    transpose_v_f16<<<(Nt * DKV) / 256, 256, 0, stream>>>(vf, vt16);

    // ---- attention, per head (GQA: kv head = h/4)
    for (int h = 0; h < 32; ++h) {
        const _Float16* qh  = q16 + h * 128;
        const _Float16* kh  = k16 + (h >> 2) * 128;
        const _Float16* vth = vt16 + (size_t)(h >> 2) * 128 * Nt;

        // scores[2048,2048] = Qh[2048,128] * Kh[2048,128]^T
        wmma_gemm<EPI_F32><<<dim3(Nt / 128, Nt / 128), 256, 0, stream>>>(
            qh, kh, nullptr, scores, 128, DM, DKV, Nt);

        softmax_causal<<<Nt, 256, 0, stream>>>(scores, p16, Nt);

        // attn[:, h*128 : (h+1)*128] = P[2048,2048] * V[2048,128]
        //   (B = V^T as [128,2048] row-major)
        wmma_gemm<EPI_F16><<<dim3(1, Nt / 128), 256, 0, stream>>>(
            p16, vth, nullptr, attn16 + h * 128, Nt, Nt, Nt, DM);
    }

    // ---- output projection + residual
    cvt(wo, (long)DM * DM);
    wmma_gemm<EPI_F32_RES><<<dim3(DM / 128, Nt / 128), 256, 0, stream>>>(
        attn16, wbuf, x, x2, DM, DM, DM, DM);

    // ---- FFN
    rmsnorm_f16<<<Nt, 256, 0, stream>>>(x2, fnw, h2_16, DM);

    cvt(wg, (long)DF * DM);
    wmma_gemm<EPI_F32><<<dim3(DF / 128, Nt / 128), 256, 0, stream>>>(
        h2_16, wbuf, nullptr, gf, DM, DM, DM, DF);
    cvt(wh, (long)DF * DM);
    wmma_gemm<EPI_F32><<<dim3(DF / 128, Nt / 128), 256, 0, stream>>>(
        h2_16, wbuf, nullptr, uf, DM, DM, DM, DF);

    swiglu_f16<<<((size_t)Nt * DF) / 256, 256, 0, stream>>>(gf, uf, ff16);

    cvt(wf, (long)DM * DF);
    wmma_gemm<EPI_F32_RES><<<dim3(DM / 128, Nt / 128), 256, 0, stream>>>(
        ff16, wbuf, x2, (float*)d_out, DF, DF, DF, DM);
}
